// DeformConv1d_90941637525679
// MI455X (gfx1250) — matrix operand order repaired and checked by reading, not compiled, not run on hardware
//
#include <hip/hip_runtime.h>

// Problem constants (match the reference)
#define B_      4
#define CIN     512
#define COUT    512
#define L_      4096
#define K_      7
#define OUT_LEN 4090          // (4096 - 6 - 1)/1 + 1
#define CK      (CIN * K_)    // 3584 reduction length
#define NKC     (CK / 32)     // 112 K-steps

// Tile config
#define MT   128              // COUT tile
#define NT   128              // OUT_LEN tile
#define KT   32               // bf16 WMMA K
#define LDP  40               // padded LDS row (halves): 80B rows, 16B-aligned, conflict-free

typedef __attribute__((ext_vector_type(16))) __bf16 v16bf;
typedef __attribute__((ext_vector_type(8)))  __bf16 v8bf;
typedef __attribute__((ext_vector_type(8)))  float  v8f;

__device__ __forceinline__ __bf16 f2bf(float f) {
    union { float f; unsigned u; } v; v.f = f;
    unsigned r = v.u + 0x7FFFu + ((v.u >> 16) & 1u);   // round-to-nearest-even
    unsigned short h = (unsigned short)(r >> 16);
    __bf16 o; __builtin_memcpy(&o, &h, 2);
    return o;
}

__launch_bounds__(256)
__global__ void deform_conv1d_wmma(const float* __restrict__ x,
                                   const float* __restrict__ offs,
                                   const float* __restrict__ w,
                                   const float* __restrict__ bias,
                                   float* __restrict__ out)
{
    // Double-buffered tiles: stage kc+1 while computing kc (one barrier per K-step)
    __shared__ __bf16 Alds[2][MT * LDP];   // weight tile: [m][kk]
    __shared__ __bf16 Blds[2][NT * LDP];   // sampled tile: [oo][kk]
    __shared__ int    u0t[NT * K_];        // floor sample index
    __shared__ float  ft [NT * K_];        // fractional weight: samp = x0 + f*(x1-x0)

    const int tid  = threadIdx.x;
    const int lane = tid & 31;
    const int wave = tid >> 5;
    const int nb = blockIdx.x, mb = blockIdx.y, b = blockIdx.z;
    const int n0 = nb * NT, m0 = mb * MT;

    // ---- Per-block interp table: T = clamp(o + k + off, o, o+6) in-range =>
    //      g0 = 1-f, g1 = f with f = T - u0 (f = 1 at the u0 = L-2 clip) ----
    for (int i = tid; i < NT * K_; i += 256) {
        int oo = i / K_, k = i % K_;
        int o  = n0 + oo;
        int u0 = 0; float f = 0.f;
        if (o < OUT_LEN) {
            float t0 = (float)o;
            float T  = t0 + (float)k + offs[(size_t)b * OUT_LEN * K_ + (size_t)o * K_ + k];
            T = fmaxf(T, t0);
            T = fminf(T, t0 + (float)(K_ - 1));
            int fu = (int)floorf(T);
            if (fu > L_ - 2) fu = L_ - 2;   // T >= o >= 0, no low clip needed
            u0 = fu;
            f  = T - (float)fu;             // in [0,1]
        }
        u0t[i] = u0; ft[i] = f;             // o >= OUT_LEN: u0=0, f=0 -> samp = x[c][0]
        // NOTE: for o >= OUT_LEN we must force samp = 0; f=0 gives samp = x0.
        // Keep f = 0 and u0 = 0 here; stage() sets the sampled value to 0 for these columns.
    }
    __syncthreads();

    v8f acc[8] = {};
    const float* xb = x + (size_t)b * CIN * L_;
    const int h = lane >> 4;           // lane half selects K sub-groups
    const int r = lane & 15;
    const bool ocheck[1] = { (n0 + NT) <= OUT_LEN };  // full tile fast-path flag

    // Staging for one K-step into buffer `buf`
    auto stage = [&](int buf, int kc) {
        const int k     = kc >> 4;          // tap 0..6
        const int cbase = (kc & 15) << 5;   // channel base 0..480
        // A: Alds[m][kk] = bf16(weight[m0+m][cbase+kk][k])
        #pragma unroll
        for (int i = tid; i < MT * KT; i += 256) {
            int m = i >> 5, kk = i & 31;
            Alds[buf][m * LDP + kk] =
                f2bf(w[(size_t)(m0 + m) * CK + (size_t)(cbase + kk) * K_ + k]);
        }
        // B (fused interp): Blds[oo][kk] = bf16(x0 + f*(x1-x0)), 0 for o >= OUT_LEN
        #pragma unroll
        for (int i = tid; i < KT * NT; i += 256) {
            int kk = i >> 7, oo = i & 127;          // oo fastest -> coalesced x gathers
            int ti = oo * K_ + k;
            const float* xr = xb + (size_t)(cbase + kk) * L_;
            int   u0 = u0t[ti];
            float f  = ft[ti];
            float x0 = xr[u0], x1 = xr[u0 + 1];
            float val = fmaf(f, x1 - x0, x0);
            if (!ocheck[0] && (n0 + oo) >= OUT_LEN) val = 0.f;
            Blds[buf][oo * LDP + kk] = f2bf(val);
        }
    };

    stage(0, 0);
    __syncthreads();

    // ---- K loop: 112 chunks of 32 (reduction index = k*512 + c) ----
    for (int kc = 0; kc < NKC; ++kc) {
        const int cur = kc & 1;

        // Stage next chunk into the other buffer; overlaps with compute below.
        if (kc + 1 < NKC) stage(cur ^ 1, kc + 1);

        // B fragment (32x16, bf16): N = lane&15 (+wave strip), K = 16h + e
        const __bf16* bp = &Blds[cur][(wave * 16 + r) * LDP + h * 16];
        v8bf b0 = *(const v8bf*)(bp);
        v8bf b1 = *(const v8bf*)(bp + 8);
        v16bf bfrag = __builtin_shufflevector(b0, b1,
            0,1,2,3,4,5,6,7,8,9,10,11,12,13,14,15);

        #pragma unroll
        for (int m = 0; m < 8; ++m) {
            // A fragment (16x32, bf16): M = lane&15; K = 16h + e
            const __bf16* ap = &Alds[cur][(m * 16 + r) * LDP];
            v8bf a0 = *(const v8bf*)(ap + 16 * h);
            v8bf a1 = *(const v8bf*)(ap + 16 * h + 8);
            v16bf afrag = __builtin_shufflevector(a0, a1,
                0,1,2,3,4,5,6,7,8,9,10,11,12,13,14,15);
            acc[m] = __builtin_amdgcn_wmma_f32_16x16x32_bf16(
                false, afrag, false, bfrag, (short)0, acc[m], false, false);
        }
        __syncthreads();   // next-buffer staged AND current-buffer reads done
    }

    // ---- Epilogue: C/D layout M = v + 8h, N = lane&15; add bias, guard N tail ----
    const int o = n0 + wave * 16 + r;
    if (o < OUT_LEN) {
        #pragma unroll
        for (int m = 0; m < 8; ++m) {
            #pragma unroll
            for (int vv = 0; vv < 8; ++vv) {
                int d = m0 + m * 16 + 8 * h + vv;
                out[(size_t)b * COUT * OUT_LEN + (size_t)d * OUT_LEN + o] =
                    acc[m][vv] + bias[d];
            }
        }
    }
}

extern "C" void kernel_launch(void* const* d_in, const int* in_sizes, int n_in,
                              void* d_out, int out_size, void* d_ws, size_t ws_size,
                              hipStream_t stream) {
    const float* x    = (const float*)d_in[0];
    const float* offs = (const float*)d_in[1];
    const float* w    = (const float*)d_in[2];
    const float* bias = (const float*)d_in[3];
    float* out        = (float*)d_out;

    dim3 grid((OUT_LEN + NT - 1) / NT, COUT / MT, B_);
    dim3 block(256, 1, 1);
    hipLaunchKernelGGL(deform_conv1d_wmma, grid, block, 0, stream,
                       x, offs, w, bias, out);
}
